// MultiHeadAttention_6949257085301
// MI455X (gfx1250) — matrix operand order repaired and checked by reading, not compiled, not run on hardware
//
#include <hip/hip_runtime.h>
#include <hip/hip_bf16.h>

// ---------------------------------------------------------------------------
// MHA forward for MI455X (gfx1250, wave32, WMMA 16x16x32 f16->f32)
// Pipeline:
//   cvt        : f32->f16 for Wq/Wk/Wv/Wo (once) and query/key/value
//   gemm<EPI=0>: Q = x @ Wq^T + bq  -> f16 [B,H,S,64]
//   gemm<EPI=0>: K = x @ Wk^T + bk  -> f16 [B,H,S,64]
//   gemm<EPI=1>: V = x @ Wv^T + bv  -> f16 [B,H,64,S]  (transposed)
//   attn       : causal flash attention -> f16 ctx [B*S, 1024]
//   gemm<EPI=2>: out = ctx @ Wo^T + bo -> f32 [B*S, 1024]
// GEMMs are all-f16 with double-buffered GLOBAL_LOAD_ASYNC_TO_LDS_B128
// staging (ASYNCcnt-synchronized), so the K-loop is pure WMMA + async DMA.
// ---------------------------------------------------------------------------

typedef __attribute__((ext_vector_type(16))) _Float16 v16h;
typedef __attribute__((ext_vector_type(8)))  _Float16 v8h;
typedef __attribute__((ext_vector_type(8)))  float    v8f;

constexpr int Bv = 4;
constexpr int Sv = 2048;
constexpr int Dm = 1024;
constexpr int Hh = 16;
constexpr int Dk = 64;

__device__ __forceinline__ v16h cat16(v8h lo, v8h hi) {
  v16h r;
#pragma unroll
  for (int i = 0; i < 8; ++i) { r[i] = lo[i]; r[i + 8] = hi[i]; }
  return r;
}

__device__ __forceinline__ v8f wmma_f16(v16h a, v16h b, v8f c) {
  return __builtin_amdgcn_wmma_f32_16x16x32_f16(
      false, a, false, b, (short)0, c, false, false);
}

__device__ __forceinline__ v8h cvt8(const float* p) {
  const float4 a = *(const float4*)p;
  const float4 b = *(const float4*)(p + 4);
  v8h h;
  h[0] = (_Float16)a.x; h[1] = (_Float16)a.y; h[2] = (_Float16)a.z; h[3] = (_Float16)a.w;
  h[4] = (_Float16)b.x; h[5] = (_Float16)b.y; h[6] = (_Float16)b.z; h[7] = (_Float16)b.w;
  return h;
}

// ---------------------------------------------------------------------------
// Async global->LDS copy of 16 bytes per lane (GLOBAL_LOAD_ASYNC_TO_LDS_B128:
// VDST = per-lane LDS byte address, VADDR = per-lane 64-bit global address).
// LDS aperture rule: generic pointer low 32 bits == LDS offset.
// Tracked by ASYNCcnt; pair with s_wait_asynccnt + barrier.
// ---------------------------------------------------------------------------
__device__ __forceinline__ void async_copy16(_Float16* ldst, const _Float16* gsrc) {
  const unsigned int laddr = (unsigned int)(unsigned long long)ldst;
  const unsigned long long gaddr = (unsigned long long)gsrc;
  asm volatile("global_load_async_to_lds_b128 %0, %1, off"
               :
               : "v"(laddr), "v"(gaddr)
               : "memory");
}

__device__ __forceinline__ void wait_async_lds() {
#if __has_builtin(__builtin_amdgcn_s_wait_asynccnt)
  __builtin_amdgcn_s_wait_asynccnt(0);
#else
  asm volatile("s_wait_asynccnt 0" ::: "memory");
#endif
}

// ---------------------------------------------------------------------------
// f32 -> f16 elementwise convert, 8 elements/thread.
// ---------------------------------------------------------------------------
__global__ __launch_bounds__(256)
void cvt_kernel(const float* __restrict__ src, _Float16* __restrict__ dst) {
  const size_t i = ((size_t)blockIdx.x * 256 + threadIdx.x) * 8;
  *(v8h*)(dst + i) = cvt8(src + i);
}

// ---------------------------------------------------------------------------
// GEMM: Out[M=8192, N=1024] = A[M,K=1024]_f16 @ W16[N,K]_f16^T + bias[N]
// 256 threads = 8 waves; block tile 128x128; K-step 32; double-buffered LDS
// filled by async b128 copies (4 async instructions per wave per tile).
// Each wave computes a 64x32 patch = 4x2 WMMA tiles.
// EPI: 0 -> f16 [B,H,S,64]; 1 -> f16 [B,H,64,S]; 2 -> f32 [M,N]
// ---------------------------------------------------------------------------
template <int EPI>
__global__ __launch_bounds__(256)
void gemm_kernel(const _Float16* __restrict__ A,
                 const _Float16* __restrict__ W,
                 const float* __restrict__ bias,
                 void* __restrict__ Out) {
  constexpr int Kd  = Dm;
  constexpr int LST = 40;  // padded LDS row stride (halves)
  __shared__ _Float16 sA[2][128 * LST];
  __shared__ _Float16 sB[2][128 * LST];

  const int tid  = threadIdx.x;
  const int lane = tid & 31;
  const int wave = tid >> 5;
  const int lm   = lane & 15;
  const int lh   = lane >> 4;
  const int mBlock = blockIdx.y * 128;
  const int nBlock = blockIdx.x * 128;
  const int wm = (wave & 1) * 64;
  const int wn = (wave >> 1) * 32;

  const int ldRow = tid >> 1;        // 0..127
  const int ldK   = (tid & 1) * 16;  // 0 or 16

  const _Float16* Arow = A + (size_t)(mBlock + ldRow) * Kd + ldK;
  const _Float16* Wrow = W + (size_t)(nBlock + ldRow) * Kd + ldK;

  auto stage = [&](int k0, int b) {
    async_copy16(&sA[b][ldRow * LST + ldK],     Arow + k0);
    async_copy16(&sA[b][ldRow * LST + ldK + 8], Arow + k0 + 8);
    async_copy16(&sB[b][ldRow * LST + ldK],     Wrow + k0);
    async_copy16(&sB[b][ldRow * LST + ldK + 8], Wrow + k0 + 8);
  };

  v8f acc[4][2];
#pragma unroll
  for (int mt = 0; mt < 4; ++mt)
#pragma unroll
    for (int nt = 0; nt < 2; ++nt) acc[mt][nt] = (v8f)0.0f;

  stage(0, 0);  // prologue: tile 0 in flight
  int buf = 0;

  for (int k0 = 0; k0 < Kd; k0 += 32) {
    wait_async_lds();   // this wave's copies for tile k0 landed
    __syncthreads();    // ...and everyone else's too
    if (k0 + 32 < Kd) stage(k0 + 32, buf ^ 1);  // next tile overlaps compute

    v16h af[4];
#pragma unroll
    for (int mt = 0; mt < 4; ++mt) {
      const int row = wm + mt * 16 + lm;
      af[mt] = cat16(*(const v8h*)&sA[buf][row * LST + lh * 16],
                     *(const v8h*)&sA[buf][row * LST + lh * 16 + 8]);
    }
    v16h bf[2];
#pragma unroll
    for (int nt = 0; nt < 2; ++nt) {
      const int nrow = wn + nt * 16 + lm;
      bf[nt] = cat16(*(const v8h*)&sB[buf][nrow * LST + lh * 16],
                     *(const v8h*)&sB[buf][nrow * LST + lh * 16 + 8]);
    }
#pragma unroll
    for (int mt = 0; mt < 4; ++mt)
#pragma unroll
      for (int nt = 0; nt < 2; ++nt)
        acc[mt][nt] = wmma_f16(af[mt], bf[nt], acc[mt][nt]);

    __syncthreads();    // all reads of sA/sB[buf] done before it is refilled
    buf ^= 1;
  }

  // C layout: reg r -> row r (lanes<16) / r+8 (lanes>=16); col = lm
#pragma unroll
  for (int mt = 0; mt < 4; ++mt) {
#pragma unroll
    for (int nt = 0; nt < 2; ++nt) {
      const int n = nBlock + wn + nt * 16 + lm;
      const float bia = bias[n];
#pragma unroll
      for (int r = 0; r < 8; ++r) {
        const int m = mBlock + wm + mt * 16 + r + 8 * lh;
        const float v = acc[mt][nt][r] + bia;
        if (EPI == 2) {
          ((float*)Out)[(size_t)m * Dm + n] = v;
        } else {
          const int b = m >> 11, s = m & (Sv - 1);
          const int h = n >> 6, d = n & (Dk - 1);
          const size_t idx =
              (EPI == 0) ? ((((size_t)b * Hh + h) * Sv + s) * Dk + d)
                         : ((((size_t)b * Hh + h) * Dk + d) * Sv + s);
          ((_Float16*)Out)[idx] = (_Float16)v;
        }
      }
    }
  }
}

// ---------------------------------------------------------------------------
// Flash attention, causal. Grid: (S/128, B*H), block 256 (8 waves).
// Wave owns 16 query rows, streams kv in blocks of 32; Q/K fragments and
// V^T fragments load directly from global as b128 pairs; P goes through a
// per-wave LDS tile for the C-layout -> A-layout conversion.
// ---------------------------------------------------------------------------
__global__ __launch_bounds__(256)
void attn_kernel(const _Float16* __restrict__ Q,
                 const _Float16* __restrict__ Kt,
                 const _Float16* __restrict__ Vt,
                 _Float16* __restrict__ ctx) {
  constexpr int PST = 40;
  __shared__ _Float16 sP[8 * 16 * PST];

  const int tid  = threadIdx.x;
  const int lane = tid & 31;
  const int wave = tid >> 5;
  const int lm = lane & 15, lh = lane >> 4;
  const int bh = blockIdx.y;
  const int q0 = blockIdx.x * 128 + wave * 16;

  const _Float16* Qb = Q  + (size_t)bh * Sv * Dk;
  const _Float16* Kb = Kt + (size_t)bh * Sv * Dk;
  const _Float16* Vb = Vt + (size_t)bh * Dk * Sv;

  v16h qf[2];
#pragma unroll
  for (int dc = 0; dc < 2; ++dc) {
    const _Float16* qp = Qb + (size_t)(q0 + lm) * Dk + dc * 32 + lh * 16;
    qf[dc] = cat16(*(const v8h*)qp, *(const v8h*)(qp + 8));
  }

  float mI[8], lI[8];
  v8f accO[4];
#pragma unroll
  for (int r = 0; r < 8; ++r) { mI[r] = -1e30f; lI[r] = 0.0f; }
#pragma unroll
  for (int dt = 0; dt < 4; ++dt) accO[dt] = (v8f)0.0f;

  _Float16* pBase = &sP[wave * 16 * PST];

  for (int kv0 = 0; kv0 <= q0 + 15; kv0 += 32) {
    v8f st[2] = {(v8f)0.0f, (v8f)0.0f};
#pragma unroll
    for (int nt = 0; nt < 2; ++nt) {
#pragma unroll
      for (int dc = 0; dc < 2; ++dc) {
        const int kvp = kv0 + nt * 16 + lm;
        const _Float16* kp = Kb + (size_t)kvp * Dk + dc * 32 + lh * 16;
        v16h bf = cat16(*(const v8h*)kp, *(const v8h*)(kp + 8));
        st[nt] = wmma_f16(qf[dc], bf, st[nt]);
      }
    }
#pragma unroll
    for (int nt = 0; nt < 2; ++nt) {
#pragma unroll
      for (int r = 0; r < 8; ++r) {
        float s = st[nt][r] * 0.125f;  // 1/sqrt(64)
        const int col = kv0 + nt * 16 + lm;
        const int row = q0 + r + 8 * lh;
        st[nt][r] = (col > row) ? -1e30f : s;
      }
    }
    float al[8];
#pragma unroll
    for (int r = 0; r < 8; ++r) {
      float mx = fmaxf(st[0][r], st[1][r]);
#pragma unroll
      for (int off = 1; off < 16; off <<= 1)
        mx = fmaxf(mx, __shfl_xor(mx, off, 32));
      const float nm = fmaxf(mI[r], mx);
      al[r] = __expf(mI[r] - nm);
      const float p0 = __expf(st[0][r] - nm);
      const float p1 = __expf(st[1][r] - nm);
      st[0][r] = p0; st[1][r] = p1;
      float sum = p0 + p1;
#pragma unroll
      for (int off = 1; off < 16; off <<= 1)
        sum += __shfl_xor(sum, off, 32);
      lI[r] = lI[r] * al[r] + sum;
      mI[r] = nm;
    }
#pragma unroll
    for (int dt = 0; dt < 4; ++dt)
#pragma unroll
      for (int r = 0; r < 8; ++r) accO[dt][r] *= al[r];

#pragma unroll
    for (int nt = 0; nt < 2; ++nt)
#pragma unroll
      for (int r = 0; r < 8; ++r)
        pBase[(r + 8 * lh) * PST + nt * 16 + lm] = (_Float16)st[nt][r];
    v16h pf = cat16(*(const v8h*)&pBase[lm * PST + lh * 16],
                    *(const v8h*)&pBase[lm * PST + lh * 16 + 8]);

#pragma unroll
    for (int dt = 0; dt < 4; ++dt) {
      const _Float16* vp = Vb + (size_t)(dt * 16 + lm) * Sv + kv0 + lh * 16;
      v16h vf = cat16(*(const v8h*)vp, *(const v8h*)(vp + 8));
      accO[dt] = wmma_f16(pf, vf, accO[dt]);
    }
  }

  const int b = bh >> 4, h = bh & 15;
#pragma unroll
  for (int dt = 0; dt < 4; ++dt) {
#pragma unroll
    for (int r = 0; r < 8; ++r) {
      const int row = q0 + r + 8 * lh;
      const int j = h * Dk + dt * 16 + lm;
      const float o = accO[dt][r] / lI[r];
      ctx[(size_t)(b * Sv + row) * Dm + j] = (_Float16)o;
    }
  }
}

// ---------------------------------------------------------------------------
extern "C" void kernel_launch(void* const* d_in, const int* in_sizes, int n_in,
                              void* d_out, int out_size, void* d_ws, size_t ws_size,
                              hipStream_t stream) {
  const float* query = (const float*)d_in[0];
  const float* key   = (const float*)d_in[1];
  const float* value = (const float*)d_in[2];
  // d_in[3] = mask: causal triu, applied analytically in attn_kernel
  const float* Wq = (const float*)d_in[4];
  const float* bq = (const float*)d_in[5];
  const float* Wk = (const float*)d_in[6];
  const float* bk = (const float*)d_in[7];
  const float* Wv = (const float*)d_in[8];
  const float* bv = (const float*)d_in[9];
  const float* Wo = (const float*)d_in[10];
  const float* bo = (const float*)d_in[11];

  char* w = (char*)d_ws;
  _Float16* wsQ  = (_Float16*)(w);                        // 16 MB [B,H,S,64]
  _Float16* wsK  = (_Float16*)(w + ((size_t)16 << 20));   // 16 MB [B,H,S,64]
  _Float16* wsV  = (_Float16*)(w + ((size_t)32 << 20));   // 16 MB [B,H,64,S]
  _Float16* wsC  = (_Float16*)(w + ((size_t)48 << 20));   // 16 MB [B*S, 1024]
  _Float16* wsX  = (_Float16*)(w + ((size_t)64 << 20));   // 16 MB f16 activations
  _Float16* wsWq = (_Float16*)(w + ((size_t)80 << 20));   // 2 MB each
  _Float16* wsWk = (_Float16*)(w + ((size_t)82 << 20));
  _Float16* wsWv = (_Float16*)(w + ((size_t)84 << 20));
  _Float16* wsWo = (_Float16*)(w + ((size_t)86 << 20));

  const int nAct = Bv * Sv * Dm;   // 8M elements
  const int nWgt = Dm * Dm;        // 1M elements
  const dim3 gg(Dm / 128, (Bv * Sv) / 128);  // (8, 64)

  cvt_kernel<<<nWgt / 2048, 256, 0, stream>>>(Wq, wsWq);
  cvt_kernel<<<nWgt / 2048, 256, 0, stream>>>(Wk, wsWk);
  cvt_kernel<<<nWgt / 2048, 256, 0, stream>>>(Wv, wsWv);
  cvt_kernel<<<nWgt / 2048, 256, 0, stream>>>(Wo, wsWo);

  cvt_kernel<<<nAct / 2048, 256, 0, stream>>>(query, wsX);
  gemm_kernel<0><<<gg, 256, 0, stream>>>(wsX, wsWq, bq, wsQ);

  cvt_kernel<<<nAct / 2048, 256, 0, stream>>>(key, wsX);
  gemm_kernel<0><<<gg, 256, 0, stream>>>(wsX, wsWk, bk, wsK);

  cvt_kernel<<<nAct / 2048, 256, 0, stream>>>(value, wsX);
  gemm_kernel<1><<<gg, 256, 0, stream>>>(wsX, wsWv, bv, wsV);

  attn_kernel<<<dim3(Sv / 128, Bv * Hh), 256, 0, stream>>>(wsQ, wsK, wsV, wsC);

  gemm_kernel<2><<<gg, 256, 0, stream>>>(wsC, wsWo, bo, (float*)d_out);
}
